// EncoderRNN_16578573762608
// MI455X (gfx1250) — matrix operand order repaired and checked by reading, not compiled, not run on hardware
//
#include <hip/hip_runtime.h>
#include <hip/hip_bf16.h>
#include <stdint.h>

// Problem constants (from reference)
#define VOCAB 64
#define E_    256
#define H_    1024
#define B_    128
#define T_    512
#define KTOT  1280          // E + H  (fused [x_t | h] @ [Wx ; Wh])
#define N4H   4096          // 4*H gate columns (gate-interleaved)
#define RATE  0.2f

#if __has_builtin(__builtin_amdgcn_sched_barrier)
#define SCHED_FENCE() __builtin_amdgcn_sched_barrier(0)
#else
#define SCHED_FENCE() asm volatile("" ::: "memory")
#endif

typedef __attribute__((ext_vector_type(16))) __bf16 v16bf;
typedef __attribute__((ext_vector_type(8)))  float  v8f;

union Frag { v16bf v; uint4 q[2]; };

__device__ __forceinline__ uint32_t hash_u32(uint32_t x, uint32_t salt) {
    uint32_t s = x * 0x9E3779B9u + salt;
    s ^= s >> 16; s *= 0x85EBCA6Bu; s ^= s >> 13; s *= 0xC2B2AE35u; s ^= s >> 16;
    return s;
}

// ---------------------------------------------------------------------------
// Prep 1: gate-interleave + bf16-pack weights.
// Wg[dir][k][n]  (row-major, k in [0,1280), n in [0,4096))
//   n = 4*j + g  maps to original column g*H + j   (Keras gate order i,f,g,o)
//   k < 256 -> Wx row k ; else Wh row k-256
// ---------------------------------------------------------------------------
__global__ void prep_weights(const float* __restrict__ Wx_f, const float* __restrict__ Wh_f,
                             const float* __restrict__ b_f,
                             const float* __restrict__ Wx_b, const float* __restrict__ Wh_b,
                             const float* __restrict__ b_b,
                             __bf16* __restrict__ Wg, float* __restrict__ bias_il)
{
    int row = blockIdx.x;            // 0 .. 2*KTOT-1
    int dir = row / KTOT;
    int k   = row % KTOT;
    const float* Wx = dir ? Wx_b : Wx_f;
    const float* Wh = dir ? Wh_b : Wh_f;
    const float* src = (k < E_) ? (Wx + (size_t)k * N4H)
                                : (Wh + (size_t)(k - E_) * N4H);
    __bf16* dst = Wg + ((size_t)dir * KTOT + k) * N4H;
    for (int n = threadIdx.x; n < N4H; n += blockDim.x) {
        int col = (n & 3) * H_ + (n >> 2);
        dst[n] = (__bf16)src[col];
    }
    if (k == 0) {
        const float* bsrc = dir ? b_b : b_f;
        for (int n = threadIdx.x; n < N4H; n += blockDim.x)
            bias_il[dir * N4H + n] = bsrc[(n & 3) * H_ + (n >> 2)];
    }
}

// ---------------------------------------------------------------------------
// Prep 2: embedding lookup + input dropout + length mask -> bf16 x [B,T,E]
// ---------------------------------------------------------------------------
__global__ void prep_x(const int* __restrict__ tokens, const int* __restrict__ lengths,
                       const float* __restrict__ emb, __bf16* __restrict__ xbf)
{
    int bt = blockIdx.x;             // b*T + t
    int b  = bt >> 9;
    int t  = bt & (T_ - 1);
    int e  = threadIdx.x;            // 256 threads == E
    int tok = tokens[bt];
    float v = emb[tok * E_ + e];
    uint32_t s = hash_u32((uint32_t)(bt * E_ + e), 0x01234567u);
    const uint32_t thresh = (uint32_t)(RATE * 4294967296.0);
    v = (s >= thresh) ? v * (1.0f / (1.0f - RATE)) : 0.0f;
    if (t >= lengths[b]) v = 0.0f;
    xbf[(size_t)bt * E_ + e] = (__bf16)v;
}

// ---------------------------------------------------------------------------
// One LSTM timestep, both directions.
// grid = (16 n-tiles of 256 cols, 4 batch strips of 32 rows, 2 dirs), 256 thr.
// z-tile [32 x 256] = [x_t|h] @ Wg-slice + bias via v_wmma_f32_16x16x32_bf16.
// K loop is software-pipelined with sched_barriers so the post-RA scheduler
// cannot sink block i+1's loads below block i's WMMAs: 2 k-blocks stay in
// flight, WMMA(i) depends only on block i's loads.
// ---------------------------------------------------------------------------
__global__ void __launch_bounds__(256) lstm_step(
    const __bf16* __restrict__ xbf,   // [B,T,E] bf16
    const __bf16* __restrict__ Wg,    // [2,1280,4096] bf16, gate-interleaved
    const float*  __restrict__ bias,  // [2,4096]
    const int*    __restrict__ lengths,
    const __bf16* __restrict__ hin,   // [2,B,H] bf16 (read)
    __bf16*       __restrict__ hout,  // [2,B,H] bf16 (write)
    float*        __restrict__ cbuf,  // [2,B,H] f32 (owned r/w)
    float*        __restrict__ out,   // [B,T,H] f32 (zero-init; += here)
    int step)
{
    __shared__ float zt[32][260];     // +4 pad: dodge bank conflicts

    const int dir    = blockIdx.z;
    const int t      = dir ? (T_ - 1 - step) : step;
    const int lane   = threadIdx.x & 31;
    const int wave   = threadIdx.x >> 5;
    const int mstrip = wave & 1;              // 16-row strip within 32
    const int nwave  = (wave >> 1) * 64;      // 64-col slice within 256
    const int nblk   = blockIdx.x * 256;      // col base in [0,4096)
    const int bbase  = blockIdx.y * 32;       // batch base

    // Uniform (scalar) bases; all per-lane addressing via 32-bit byte offsets.
    const char* wbase = (const char*)(Wg + (size_t)dir * KTOT * N4H + nblk + nwave);
    const char* xbase = (const char*)xbf;
    const char* hbase = (const char*)(hin + (size_t)dir * B_ * H_);

    // --- accumulators, bias folded in (C lane L holds col n0+(L&15)) ---
    v8f acc[4];
#pragma unroll
    for (int tn = 0; tn < 4; ++tn) {
        float bv = bias[dir * N4H + nblk + nwave + tn * 16 + (lane & 15)];
        v8f a;
#pragma unroll
        for (int r = 0; r < 8; ++r) a[r] = bv;
        acc[tn] = a;
    }

    // --- A fragment addressing (ISA 16-bit A 16x32 layout) ---
    const int arow  = bbase + mstrip * 16 + (lane & 15);  // batch row
    const int akoff = (lane >> 4) * 8;                    // 0 or 8
    const uint32_t xoff = ((uint32_t)arow * T_ + (uint32_t)t) * (E_ * 2); // row byte offset
    const uint32_t hoff = (uint32_t)arow * (H_ * 2);
    const uint32_t brow = (uint32_t)(lane & 15) * 2;      // B: lane <-> N col

    // Load one 32-wide k-block: A frag + 4 B frags (2 x b128 + 64 x b16).
    auto load_block = [&](int k0, Frag& a, Frag* bf) {
        int ka0 = k0 + akoff;
        int ka1 = ka0 + 16;
        const char* p0 = (ka0 < E_) ? (xbase + xoff + (uint32_t)ka0 * 2)
                                    : (hbase + hoff + (uint32_t)(ka0 - E_) * 2);
        const char* p1 = (ka1 < E_) ? (xbase + xoff + (uint32_t)ka1 * 2)
                                    : (hbase + hoff + (uint32_t)(ka1 - E_) * 2);
        a.q[0] = *(const uint4*)p0;
        a.q[1] = *(const uint4*)p1;
        uint32_t bo = brow + (uint32_t)k0 * (N4H * 2);
        if (k0 + 64 < KTOT)            // warm L0 two blocks ahead (L2-resident)
            __builtin_prefetch((const void*)(wbase + bo + (uint32_t)64 * (N4H * 2)), 0, 3);
#pragma unroll
        for (int tn = 0; tn < 4; ++tn) {
#pragma unroll
            for (int s = 0; s < 16; ++s) {
                uint32_t kk = (uint32_t)(akoff + (s & 7) + ((s >> 3) * 16));
                bf[tn].v[s] = *(const __bf16*)(wbase + bo + kk * (N4H * 2) + tn * 32);
            }
        }
    };
    auto mm4 = [&](const Frag& a, const Frag* bf) {
#pragma unroll
        for (int tn = 0; tn < 4; ++tn)
            acc[tn] = __builtin_amdgcn_wmma_f32_16x16x32_bf16(
                false, a.v, false, bf[tn].v, (short)0, acc[tn], false, false);
    };

    // --- software-pipelined K loop: 40 blocks of 32, double-buffered.
    // sched_barriers pin program order: loads(i+1) must issue before wmma(i).
    Frag a0, a1;
    Frag b0[4], b1[4];
    load_block(0, a0, b0);
    int k0 = 0;
#pragma unroll 1
    for (; k0 < KTOT - 64; k0 += 64) {
        load_block(k0 + 32, a1, b1);      // next block's 66 loads in flight
        SCHED_FENCE();
        mm4(a0, b0);                      // waits on a0/b0 only, overlaps b1 loads
        load_block(k0 + 64, a0, b0);
        SCHED_FENCE();
        mm4(a1, b1);
    }
    // tail: k0 == KTOT-64, two blocks left
    load_block(k0 + 32, a1, b1);
    SCHED_FENCE();
    mm4(a0, b0);
    SCHED_FENCE();
    mm4(a1, b1);

    // --- spill z tile to LDS (C layout: VGPR r -> row, lane&15 -> col) ---
#pragma unroll
    for (int tn = 0; tn < 4; ++tn) {
        int col   = nwave + tn * 16 + (lane & 15);
        int rbase = mstrip * 16 + ((lane & 16) ? 8 : 0);
#pragma unroll
        for (int r = 0; r < 8; ++r)
            zt[rbase + r][col] = acc[tn][r];
    }
    __syncthreads();

    // --- gate nonlinearity; cols 4j..4j+3 = i,f,g,o of hidden unit j ---
    for (int u = threadIdx.x; u < 32 * 64; u += 256) {
        int row   = u >> 6;
        int jj    = u & 63;
        int b     = bbase + row;
        int jglob = blockIdx.x * 64 + jj;        // hidden index in [0,1024)
        size_t hidx = (size_t)dir * B_ * H_ + (size_t)b * H_ + jglob;
        if (t < lengths[b]) {
            int nl = jj * 4;
            float zi = zt[row][nl + 0];
            float zf = zt[row][nl + 1];
            float zg = zt[row][nl + 2];
            float zo = zt[row][nl + 3];
            float ig = 1.0f / (1.0f + __expf(-zi));
            float fg = 1.0f / (1.0f + __expf(-zf));
            float gg = tanhf(zg);
            float og = 1.0f / (1.0f + __expf(-zo));
            float cn = fg * cbuf[hidx] + ig * gg;
            float hn = og * tanhf(cn);
            cbuf[hidx] = cn;
            hout[hidx] = (__bf16)hn;
            out[((size_t)b * T_ + t) * H_ + jglob] += hn;   // fwd/bwd: disjoint launches
        } else {
            hout[hidx] = hin[hidx];                          // carry frozen state
        }
    }
}

// ---------------------------------------------------------------------------
// Output dropout + mask (out already holds fwd+bwd sum; masked rows are 0)
// ---------------------------------------------------------------------------
__global__ void finalize(float* __restrict__ out, const int* __restrict__ lengths)
{
    int bt = blockIdx.x;
    int b  = bt >> 9;
    int t  = bt & (T_ - 1);
    bool m = (t < lengths[b]);
    const uint32_t thresh = (uint32_t)(RATE * 4294967296.0);
    for (int j = threadIdx.x; j < H_; j += blockDim.x) {
        size_t idx = (size_t)bt * H_ + j;
        float v = out[idx];
        uint32_t s = hash_u32((uint32_t)idx, 0xDEADBEEFu);
        out[idx] = (m && s >= thresh) ? v * (1.0f / (1.0f - RATE)) : 0.0f;
    }
}

// ---------------------------------------------------------------------------
extern "C" void kernel_launch(void* const* d_in, const int* in_sizes, int n_in,
                              void* d_out, int out_size, void* d_ws, size_t ws_size,
                              hipStream_t stream)
{
    const int*   tokens  = (const int*)  d_in[0];
    const int*   lengths = (const int*)  d_in[1];
    const float* emb     = (const float*)d_in[2];
    const float* Wx_f    = (const float*)d_in[3];
    const float* Wh_f    = (const float*)d_in[4];
    const float* b_f     = (const float*)d_in[5];
    const float* Wx_b    = (const float*)d_in[6];
    const float* Wh_b    = (const float*)d_in[7];
    const float* b_b     = (const float*)d_in[8];
    float* out = (float*)d_out;

    // Workspace carve (~56 MB total)
    char* ws = (char*)d_ws;
    size_t off = 0;
    auto carve = [&](size_t bytes) {
        char* p = ws + off;
        off = (off + bytes + 255) & ~(size_t)255;
        return p;
    };
    __bf16* Wg   = (__bf16*)carve((size_t)2 * KTOT * N4H * sizeof(__bf16)); // 20 MB
    float*  bias = (float*) carve((size_t)2 * N4H * sizeof(float));
    __bf16* xbf  = (__bf16*)carve((size_t)B_ * T_ * E_ * sizeof(__bf16));   // 32 MB
    __bf16* hbuf = (__bf16*)carve((size_t)2 * 2 * B_ * H_ * sizeof(__bf16)); // ping-pong x dir
    float*  cbuf = (float*) carve((size_t)2 * B_ * H_ * sizeof(float));

    hipMemsetAsync(hbuf, 0, (size_t)2 * 2 * B_ * H_ * sizeof(__bf16), stream);
    hipMemsetAsync(cbuf, 0, (size_t)2 * B_ * H_ * sizeof(float), stream);
    hipMemsetAsync(out,  0, (size_t)B_ * T_ * H_ * sizeof(float), stream);

    prep_weights<<<2 * KTOT, 256, 0, stream>>>(Wx_f, Wh_f, b_f, Wx_b, Wh_b, b_b, Wg, bias);
    prep_x<<<B_ * T_, 256, 0, stream>>>(tokens, lengths, emb, xbf);

    const size_t hstride = (size_t)2 * B_ * H_;   // one ping-pong buffer (both dirs)
    for (int s = 0; s < T_; ++s) {
        const __bf16* hin = hbuf + (size_t)(s & 1) * hstride;
        __bf16* hout      = hbuf + (size_t)((s & 1) ^ 1) * hstride;
        dim3 grid(16, 4, 2);
        lstm_step<<<grid, dim3(256), 0, stream>>>(xbf, Wg, bias, lengths,
                                                  hin, hout, cbuf, out, s);
    }
    finalize<<<B_ * T_, 256, 0, stream>>>(out, lengths);
}
